// Net_8607114461800
// MI455X (gfx1250) — matrix operand order repaired and checked by reading, not compiled, not run on hardware
//
#include <hip/hip_runtime.h>

// ---------------- problem dims ----------------
#define NDRUG 20000
#define NGENE 4264
#define NTOT  24264
#define NEDGE 100000
#define NREL  5
#define INDIM 1613
#define DIM1  1340
#define DIM2  920
#define DIM3  740

// ---------------- padded dims (M,N -> x128 ; K -> x32) ----------------
#define MP   24320   // 190*128 >= 24264
#define K0P  1632    // 51*32  >= 1613
#define D1P  1408    // 11*128 >= 1340
#define D2P  1024    // 8*128  >= 920
#define D3P  768     // 6*128  >= 740

typedef __attribute__((ext_vector_type(16))) __bf16        v16bf;
typedef __attribute__((ext_vector_type(8)))  float         v8f;
typedef __attribute__((ext_vector_type(4)))  unsigned int  v4u;

union FragU { v4u u[2]; v16bf v; };

__device__ __forceinline__ unsigned short f2bf(float f) {
  unsigned u = __float_as_uint(f);
  u += 0x7FFFu + ((u >> 16) & 1u);   // round-to-nearest-even
  return (unsigned short)(u >> 16);
}

// ---- concat [x ; gene_emb] -> padded bf16 activation matrix [MP x K0P] ----
__global__ void k_concat_bf16(const float* __restrict__ x, const float* __restrict__ g,
                              unsigned short* __restrict__ A) {
  size_t i = (size_t)blockIdx.x * blockDim.x + threadIdx.x;
  const size_t total = (size_t)MP * K0P;
  if (i >= total) return;
  int row = (int)(i / K0P), col = (int)(i % K0P);
  float v = 0.f;
  if (row < NTOT && col < INDIM)
    v = (row < NDRUG) ? x[(size_t)row * INDIM + col]
                      : g[(size_t)(row - NDRUG) * INDIM + col];
  A[i] = f2bf(v);
}

// ---- relu + f32 -> bf16 (same padded shape) ----
__global__ void k_relu_bf16(const float* __restrict__ C, unsigned short* __restrict__ A,
                            size_t total) {
  size_t i = (size_t)blockIdx.x * blockDim.x + threadIdx.x;
  if (i >= total) return;
  A[i] = f2bf(fmaxf(C[i], 0.f));
}

// ---- weight [Kr x Nr] f32 -> transposed padded bf16 [Np x Kp] ----
__global__ void k_wt_bf16(const float* __restrict__ W, unsigned short* __restrict__ Wt,
                          int Kr, int Nr, int Kp, int Np) {
  size_t i = (size_t)blockIdx.x * blockDim.x + threadIdx.x;
  const size_t total = (size_t)Np * Kp;
  if (i >= total) return;
  int n = (int)(i / Kp), k = (int)(i % Kp);
  float v = (n < Nr && k < Kr) ? W[(size_t)k * Nr + n] : 0.f;
  Wt[i] = f2bf(v);
}

__global__ void k_bias_pad(const float* __restrict__ b, float* __restrict__ bp,
                           int Nr, int Np) {
  int i = blockIdx.x * blockDim.x + threadIdx.x;
  if (i >= Np) return;
  bp[i] = (i < Nr) ? b[i] : 0.f;
}

__global__ void k_zero_f32(float* __restrict__ p, int n) {
  int i = blockIdx.x * blockDim.x + threadIdx.x;
  if (i < n) p[i] = 0.f;
}

__global__ void k_count(const int* __restrict__ et, const int* __restrict__ dst,
                        float* __restrict__ cnt, int r) {
  int e = blockIdx.x * blockDim.x + threadIdx.x;
  if (e >= NEDGE) return;
  if (et[e] == r) atomicAdd(&cnt[dst[e]], 1.0f);
}

__global__ void k_inv(float* __restrict__ cnt, int n) {
  int i = blockIdx.x * blockDim.x + threadIdx.x;
  if (i < n) cnt[i] = 1.0f / fmaxf(cnt[i], 1.0f);
}

// ---- edge scatter: OUT[dst] += T[src] * inv_cnt[dst]  (relation r only) ----
__global__ void k_scatter(const float* __restrict__ T, float* __restrict__ OUT,
                          const int* __restrict__ src, const int* __restrict__ dst,
                          const int* __restrict__ et, const float* __restrict__ inv,
                          int r, int D, int Np) {
  int e = blockIdx.x;
  if (et[e] != r) return;
  int s = src[e], d = dst[e];
  float w = inv[d];
  const float* ts = T + (size_t)s * Np;
  float* od = OUT + (size_t)d * Np;
  for (int j = threadIdx.x; j < D; j += blockDim.x)
    atomicAdd(&od[j], ts[j] * w);
}

// issue one async 16B global->LDS copy (per lane); tracked by ASYNCcnt
#define ASYNC_CP(ldsaddr, gaddr)                                           \
  asm volatile("global_load_async_to_lds_b128 %0, %1, off"                 \
               :: "v"(ldsaddr),                                            \
                  "v"((unsigned long long)(size_t)(gaddr))                 \
               : "memory")

// ---- bf16 WMMA GEMM:  C[MP x Np] = A[MP x Kp] * Bt[Np x Kp]^T (+ bias) ----
// 256 threads = 8 wave32; block tile 128x128, K-step 32; wave tile 64x32.
// Tiles staged with GLOBAL_LOAD_ASYNC_TO_LDS_B128 into double-buffered LDS;
// steady-state loop is branch-free (last tile peeled).
#define LDK 40  // LDS row pitch in halfs (32 data + 8 pad for bank spread)
__global__ __launch_bounds__(256) void k_gemm_bf16(
    const unsigned short* __restrict__ A,
    const unsigned short* __restrict__ Bt,
    float* __restrict__ C,
    const float* __restrict__ bias,
    int Kp, int Np) {
  __shared__ __attribute__((aligned(16))) unsigned short sA[2][128 * LDK];
  __shared__ __attribute__((aligned(16))) unsigned short sB[2][128 * LDK];

  const int tid  = threadIdx.x;
  const int m0   = blockIdx.x * 128;
  const int n0   = blockIdx.y * 128;
  const int wave = tid >> 5;
  const int lane = tid & 31;
  const int hl   = lane >> 4;       // half-wave: selects K sub-block
  const int l16  = lane & 15;       // row/col within 16
  const int wm   = (wave >> 2) * 64;
  const int wn   = (wave & 3) * 32;

  v8f acc[4][2] = {};

  // cooperative staging: 512 x 16B chunks per matrix, 2 per thread
  const int c0 = tid, c1 = tid + 256;
  const int r0 = c0 >> 2, o0 = (c0 & 3) * 8;
  const int r1 = c1 >> 2, o1 = (c1 & 3) * 8;

  const unsigned short* gA0 = &A[(size_t)(m0 + r0) * Kp + o0];
  const unsigned short* gA1 = &A[(size_t)(m0 + r1) * Kp + o1];
  const unsigned short* gB0 = &Bt[(size_t)(n0 + r0) * Kp + o0];
  const unsigned short* gB1 = &Bt[(size_t)(n0 + r1) * Kp + o1];
  // LDS byte addresses: for flat LDS addresses, addr[31:0] == DS offset.
  unsigned lA0[2], lA1[2], lB0[2], lB1[2];
#pragma unroll
  for (int s = 0; s < 2; ++s) {
    lA0[s] = (unsigned)(size_t)&sA[s][r0 * LDK + o0];
    lA1[s] = (unsigned)(size_t)&sA[s][r1 * LDK + o1];
    lB0[s] = (unsigned)(size_t)&sB[s][r0 * LDK + o0];
    lB1[s] = (unsigned)(size_t)&sB[s][r1 * LDK + o1];
  }

  // one 16x16x32 sub-tile pass over buffer 'cur'
  auto compute_tile = [&](int cur) {
    FragU fa[4], fb[2];
#pragma unroll
    for (int i = 0; i < 4; ++i) {
      // A 16x32 bf16 frag: lane<16 -> K 0..7 & 16..23 ; lane>=16 -> K 8..15 & 24..31
      const unsigned short* p = &sA[cur][(wm + i * 16 + l16) * LDK];
      fa[i].u[0] = *(const v4u*)(p + hl * 8);
      fa[i].u[1] = *(const v4u*)(p + hl * 8 + 16);
    }
#pragma unroll
    for (int j = 0; j < 2; ++j) {
      // B 32x16 bf16 frag: lane<16 -> K 0..7 & 16..23 ; lane>=16 -> K 8..15 & 24..31 (col = l16)
      const unsigned short* p = &sB[cur][(wn + j * 16 + l16) * LDK];
      fb[j].u[0] = *(const v4u*)(p + hl * 8);
      fb[j].u[1] = *(const v4u*)(p + hl * 8 + 16);
    }
#pragma unroll
    for (int i = 0; i < 4; ++i)
#pragma unroll
      for (int j = 0; j < 2; ++j)
        acc[i][j] = __builtin_amdgcn_wmma_f32_16x16x32_bf16(
            false, fa[i].v, false, fb[j].v, (short)0, acc[i][j], false, false);
  };

  // prologue: stage tile 0 into buffer 0
  ASYNC_CP(lA0[0], gA0); ASYNC_CP(lA1[0], gA1);
  ASYNC_CP(lB0[0], gB0); ASYNC_CP(lB1[0], gB1);

  const int nk = Kp >> 5;
  // steady state: always prefetch next tile, branch-free
  for (int t = 0; t + 1 < nk; ++t) {
    const int cur = t & 1, nxt = cur ^ 1;
    const int k1 = (t + 1) * 32;
    ASYNC_CP(lA0[nxt], gA0 + k1); ASYNC_CP(lA1[nxt], gA1 + k1);
    ASYNC_CP(lB0[nxt], gB0 + k1); ASYNC_CP(lB1[nxt], gB1 + k1);
    // async loads complete in order: <=4 outstanding => tile t resident
    asm volatile("s_wait_asynccnt 0x4" ::: "memory");
    __syncthreads();
    compute_tile(cur);
    // all waves done reading 'cur' before it is refilled next iteration
    __syncthreads();
  }
  // peeled last tile: wait for everything, no trailing barrier needed
  asm volatile("s_wait_asynccnt 0x0" ::: "memory");
  __syncthreads();
  compute_tile((nk - 1) & 1);

  // epilogue: C layout -> VGPR v holds row (v + 8*hl), col = l16
#pragma unroll
  for (int i = 0; i < 4; ++i) {
#pragma unroll
    for (int j = 0; j < 2; ++j) {
      const int col = n0 + wn + j * 16 + l16;
      const float bv = bias ? bias[col] : 0.f;
      float* cp = C + (size_t)(m0 + wm + i * 16 + hl * 8) * Np + col;
#pragma unroll
      for (int v = 0; v < 8; ++v)
        cp[(size_t)v * Np] = acc[i][j][v] + bv;
    }
  }
}

// ---- head: emb = relu(H), logits = emb @ lin2_w + b, log_softmax ----
__global__ __launch_bounds__(256) void k_head(const float* __restrict__ H,
                                              const float* __restrict__ w2,
                                              const float* __restrict__ b2,
                                              float* __restrict__ out_ls,
                                              float* __restrict__ out_emb) {
  const int i = blockIdx.x;
  const int tid = threadIdx.x;
  float s0 = 0.f, s1 = 0.f;
  const float* hr = H + (size_t)i * D3P;
  float* er = out_emb + (size_t)i * DIM3;
  for (int j = tid; j < DIM3; j += 256) {
    float h = fmaxf(hr[j], 0.f);
    er[j] = h;
    s0 += h * w2[2 * j + 0];
    s1 += h * w2[2 * j + 1];
  }
  __shared__ float red0[256], red1[256];
  red0[tid] = s0; red1[tid] = s1;
  __syncthreads();
  for (int off = 128; off > 0; off >>= 1) {
    if (tid < off) { red0[tid] += red0[tid + off]; red1[tid] += red1[tid + off]; }
    __syncthreads();
  }
  if (tid == 0) {
    float l0 = red0[0] + b2[0], l1 = red1[0] + b2[1];
    float m = fmaxf(l0, l1);
    float lse = m + logf(expf(l0 - m) + expf(l1 - m));
    out_ls[2 * i + 0] = l0 - lse;
    out_ls[2 * i + 1] = l1 - lse;
  }
}

extern "C" void kernel_launch(void* const* d_in, const int* in_sizes, int n_in,
                              void* d_out, int out_size, void* d_ws, size_t ws_size,
                              hipStream_t stream) {
  (void)in_sizes; (void)n_in; (void)out_size; (void)ws_size;
  const float* x     = (const float*)d_in[0];
  const int*   eidx  = (const int*)d_in[1];
  const int*   etype = (const int*)d_in[2];
  const float* gene  = (const float*)d_in[3];
  const float* W1    = (const float*)d_in[4];
  const float* root1 = (const float*)d_in[5];
  const float* b1    = (const float*)d_in[6];
  const float* W2    = (const float*)d_in[7];
  const float* root2 = (const float*)d_in[8];
  const float* b2    = (const float*)d_in[9];
  const float* l1w   = (const float*)d_in[10];
  const float* l1b   = (const float*)d_in[11];
  const float* l2w   = (const float*)d_in[12];
  const float* l2b   = (const float*)d_in[13];
  const int* src = eidx;
  const int* dst = eidx + NEDGE;

  // workspace carve (aligned 256B)
  char* base = (char*)d_ws;
  auto carve = [&](size_t bytes) -> char* {
    char* p = base;
    base += (bytes + 255) & ~(size_t)255;
    return p;
  };
  unsigned short* A   = (unsigned short*)carve((size_t)MP * K0P * 2); // activations bf16
  float*          OUT = (float*)carve((size_t)MP * D1P * 4);          // layer output f32
  float*          T   = (float*)carve((size_t)MP * D1P * 4);          // per-relation transform
  unsigned short* Wt  = (unsigned short*)carve((size_t)D1P * K0P * 2);// transposed weight bf16
  float*          bp  = (float*)carve((size_t)D1P * 4);               // padded bias
  float*          cnt = (float*)carve((size_t)NTOT * 4);              // per-relation 1/cnt

  dim3 blk(256);
  auto g1 = [](size_t n) { return dim3((unsigned)((n + 255) / 256)); };

  auto run_layer = [&](const unsigned short* Ain, int Kp, const float* Wrel,
                       const float* root, const float* bias, int Kr, int Nr, int Np) {
    k_bias_pad<<<g1(Np), blk, 0, stream>>>(bias, bp, Nr, Np);
    k_wt_bf16<<<g1((size_t)Np * Kp), blk, 0, stream>>>(root, Wt, Kr, Nr, Kp, Np);
    dim3 gg(MP / 128, Np / 128);
    k_gemm_bf16<<<gg, blk, 0, stream>>>(Ain, Wt, OUT, bp, Kp, Np);
    for (int r = 0; r < NREL; ++r) {
      k_zero_f32<<<g1(NTOT), blk, 0, stream>>>(cnt, NTOT);
      k_count<<<g1(NEDGE), blk, 0, stream>>>(etype, dst, cnt, r);
      k_inv<<<g1(NTOT), blk, 0, stream>>>(cnt, NTOT);
      k_wt_bf16<<<g1((size_t)Np * Kp), blk, 0, stream>>>(
          Wrel + (size_t)r * Kr * Nr, Wt, Kr, Nr, Kp, Np);
      k_gemm_bf16<<<gg, blk, 0, stream>>>(Ain, Wt, T, (const float*)nullptr, Kp, Np);
      k_scatter<<<dim3(NEDGE), blk, 0, stream>>>(T, OUT, src, dst, etype, cnt, r, Nr, Np);
    }
  };

  // h0 = [x ; gene_emb] -> bf16
  k_concat_bf16<<<g1((size_t)MP * K0P), blk, 0, stream>>>(x, gene, A);
  // layer 1
  run_layer(A, K0P, W1, root1, b1, INDIM, DIM1, D1P);
  k_relu_bf16<<<g1((size_t)MP * D1P), blk, 0, stream>>>(OUT, A, (size_t)MP * D1P);
  // layer 2
  run_layer(A, D1P, W2, root2, b2, DIM1, DIM2, D2P);
  k_relu_bf16<<<g1((size_t)MP * D2P), blk, 0, stream>>>(OUT, A, (size_t)MP * D2P);
  // lin1 (pre-relu result kept in OUT; head applies relu)
  k_bias_pad<<<g1(D3P), blk, 0, stream>>>(l1b, bp, DIM3, D3P);
  k_wt_bf16<<<g1((size_t)D3P * D2P), blk, 0, stream>>>(l1w, Wt, DIM2, DIM3, D2P, D3P);
  dim3 gg3(MP / 128, D3P / 128);
  k_gemm_bf16<<<gg3, blk, 0, stream>>>(A, Wt, OUT, bp, D2P, D3P);
  // head: log_softmax (NTOT x 2) then emb (NTOT x 740), concatenated in d_out
  float* out_ls  = (float*)d_out;
  float* out_emb = out_ls + (size_t)NTOT * 2;
  k_head<<<dim3(NTOT), blk, 0, stream>>>(OUT, l2w, l2b, out_ls, out_emb);
}
